// self_attention_2164663517557
// MI455X (gfx1250) — matrix operand order repaired and checked by reading, not compiled, not run on hardware
//
#include <hip/hip_runtime.h>

typedef _Float16 half_t;
typedef __attribute__((ext_vector_type(16))) _Float16 v16h;
typedef __attribute__((ext_vector_type(8)))  float    v8f;

#define BM 128
#define BN 128
#define BK 32
#define PITCH 40   // halves per LDS row: 32 data + 8 pad = 80 bytes (16B-aligned rows)
#define BUFSTRIDE ((BM + BN) * PITCH)   // halves per LDS double-buffer slot

__device__ __forceinline__ v8f wmma16(v16h a, v16h b, v8f c) {
  return __builtin_amdgcn_wmma_f32_16x16x32_f16(false, a, false, b, (short)0, c, false, false);
}

// ---- Async tile staging: 128x32 halves (8KB), 2 x b128 per thread ----------
// src row-major [rows][ld], K contiguous; lds tile [row][PITCH].
__device__ __forceinline__ void stage_tile_async(const half_t* __restrict__ src, int ld,
                                                 int row0, int k0,
                                                 half_t* lds, int tid) {
  #pragma unroll
  for (int h = 0; h < 2; ++h) {
    int c  = tid + h * 256;        // 0..511 chunks of 8 halves
    int r  = c >> 2;               // 0..127
    int kc = (c & 3) * 8;          // 0,8,16,24
    const half_t* g = src + (size_t)(row0 + r) * ld + (k0 + kc);
    unsigned lo = (unsigned)(uintptr_t)(lds + r * PITCH + kc);
    asm volatile("global_load_async_to_lds_b128 %0, %1, off"
                 :: "v"(lo), "v"(g) : "memory");
  }
}

// ---- WMMA fragment loads (ISA 16-bit A/B VGPR layouts, wave32) -------------
// A 16x32: lanes 0-15 rows M, K {0..7} in V0..3 and {16..23} in V4..7; lanes 16-31 K +8.
__device__ __forceinline__ v16h load_frag_a(const half_t* __restrict__ lds,
                                            int row_base, int lane) {
  int row = row_base + (lane & 15);
  int kg  = (lane >> 4) * 8;
  const half_t* base = lds + row * PITCH;
  union { uint4 u[2]; v16h v; } r;
  r.u[0] = *reinterpret_cast<const uint4*>(base + kg);
  r.u[1] = *reinterpret_cast<const uint4*>(base + 16 + kg);
  return r.v;
}

// B 32x16: lanes 0-15 col N, K {0..7} in V0..3 and {16..23} in V4..7; lanes 16-31 K +8.
__device__ __forceinline__ v16h load_frag_b(const half_t* __restrict__ lds,
                                            int n_base, int lane) {
  int n  = n_base + (lane & 15);
  int kg = (lane >> 4) * 8;
  const half_t* base = lds + n * PITCH + kg;
  union { uint4 u[2]; v16h v; } r;
  r.u[0] = *reinterpret_cast<const uint4*>(base);
  r.u[1] = *reinterpret_cast<const uint4*>(base + 16);
  return r.v;
}

// ---- Core: 128x128 block, 8 waves (4x2), wave tile 32x64 = 8 WMMA/k-step ---
// Both A and B are [rows][K] row-major, K contiguous. Double-buffered async LDS.
__device__ __forceinline__ void wmma_core_async(const half_t* __restrict__ A, int lda,
                                                const half_t* __restrict__ Bm, int ldb,
                                                int K, int mb, int nb, int tid,
                                                half_t* sh, v8f acc[2][4]) {
  int lane = tid & 31;
  int wave = tid >> 5;
  int wm = wave >> 1;   // 0..3 -> 32-row group
  int wn = wave & 1;    // 0..1 -> 64-col group

  // prologue: stage k0=0 into buffer 0 (4 async instructions per wave)
  stage_tile_async(A,  lda, mb, 0, sh,              tid);
  stage_tile_async(Bm, ldb, nb, 0, sh + BM * PITCH, tid);

  int nstep = K / BK;
  int cur = 0;
  for (int s = 0; s < nstep; ++s) {
    if (s + 1 < nstep) {
      half_t* nxt = sh + (1 - cur) * BUFSTRIDE;
      stage_tile_async(A,  lda, mb, (s + 1) * BK, nxt,              tid);
      stage_tile_async(Bm, ldb, nb, (s + 1) * BK, nxt + BM * PITCH, tid);
      asm volatile("s_wait_asynccnt 4" ::: "memory");   // current buffer done
    } else {
      asm volatile("s_wait_asynccnt 0" ::: "memory");
    }
    __syncthreads();

    const half_t* Asb = sh + cur * BUFSTRIDE;
    const half_t* Bsb = Asb + BM * PITCH;
    v16h a0 = load_frag_a(Asb, wm * 32,      lane);
    v16h a1 = load_frag_a(Asb, wm * 32 + 16, lane);
    v16h b[4];
    #pragma unroll
    for (int j = 0; j < 4; ++j) b[j] = load_frag_b(Bsb, wn * 64 + 16 * j, lane);
    #pragma unroll
    for (int j = 0; j < 4; ++j) {
      acc[0][j] = wmma16(a0, b[j], acc[0][j]);
      acc[1][j] = wmma16(a1, b[j], acc[1][j]);
    }
    __syncthreads();   // protect buffer before it is overwritten next step
    cur ^= 1;
  }
}

// ---- Kernel 1: C = A[M,K] @ Wt[N,K]^T + bias; optional transposed f16 out --
template <bool OUT_T>
__global__ __launch_bounds__(256) void gemm_bias(const half_t* __restrict__ A,
                                                 const half_t* __restrict__ Wt,
                                                 const float* __restrict__ bias,
                                                 half_t* __restrict__ C,
                                                 int M, int N, int K, int Sper) {
  __shared__ half_t sh[2 * BUFSTRIDE];
  int tid = threadIdx.x;
  int mb = blockIdx.y * BM, nb = blockIdx.x * BN;
  v8f acc[2][4] = {};
  wmma_core_async(A, K, Wt, K, K, mb, nb, tid, sh, acc);
  int lane = tid & 31, wave = tid >> 5, wm = wave >> 1, wn = wave & 1;
  #pragma unroll
  for (int i = 0; i < 2; ++i) {
    int rb = mb + wm * 32 + i * 16 + (lane >> 4) * 8;
    #pragma unroll
    for (int j = 0; j < 4; ++j) {
      int col = nb + wn * 64 + j * 16 + (lane & 15);
      float bv = bias[col];
      #pragma unroll
      for (int r = 0; r < 8; ++r) {
        int row = rb + r;
        float v = acc[i][j][r] + bv;
        if (OUT_T) {   // write [b][N][Sper] (transposed per batch)
          int bi = row / Sper;
          int rl = row - bi * Sper;
          C[(size_t)bi * N * Sper + (size_t)col * Sper + rl] = (half_t)v;
        } else {
          C[(size_t)row * N + col] = (half_t)v;
        }
      }
    }
  }
}

// ---- Kernel 2: scores = Q @ K^T * scale + mask (per batch, f32 out) --------
__global__ __launch_bounds__(256) void gemm_scores(const half_t* __restrict__ Qh,
                                                   const half_t* __restrict__ Kh,
                                                   const float* __restrict__ mask,
                                                   float* __restrict__ Sc,
                                                   int S, int Dk, float scale) {
  __shared__ half_t sh[2 * BUFSTRIDE];
  int b = blockIdx.z;
  const half_t* A  = Qh + (size_t)b * S * Dk;
  const half_t* Bm = Kh + (size_t)b * S * Dk;   // [key][dk]: K-contiguous
  const float* Mp  = mask + (size_t)b * S * S;
  float* Out       = Sc + (size_t)b * S * S;
  int tid = threadIdx.x;
  int mb = blockIdx.y * BM, nb = blockIdx.x * BN;
  v8f acc[2][4] = {};
  wmma_core_async(A, Dk, Bm, Dk, Dk, mb, nb, tid, sh, acc);
  int lane = tid & 31, wave = tid >> 5, wm = wave >> 1, wn = wave & 1;
  #pragma unroll
  for (int i = 0; i < 2; ++i) {
    int rb = mb + wm * 32 + i * 16 + (lane >> 4) * 8;
    #pragma unroll
    for (int j = 0; j < 4; ++j) {
      int col = nb + wn * 64 + j * 16 + (lane & 15);
      #pragma unroll
      for (int r = 0; r < 8; ++r) {
        size_t o = (size_t)(rb + r) * S + col;
        Out[o] = acc[i][j][r] * scale + Mp[o];
      }
    }
  }
}

// ---- Kernel 3: out_f32 = P[S,S] @ Vt[Dk,S]^T (per batch) -------------------
__global__ __launch_bounds__(256) void gemm_pv(const half_t* __restrict__ P,
                                               const half_t* __restrict__ Vt,
                                               float* __restrict__ Out,
                                               int S, int Dk) {
  __shared__ half_t sh[2 * BUFSTRIDE];
  int b = blockIdx.z;
  const half_t* A  = P  + (size_t)b * S * S;     // [q][key], key contiguous
  const half_t* Bm = Vt + (size_t)b * Dk * S;    // [dk][key], key contiguous
  float* O         = Out + (size_t)b * S * Dk;
  int tid = threadIdx.x;
  int mb = blockIdx.y * BM, nb = blockIdx.x * BN;
  v8f acc[2][4] = {};
  wmma_core_async(A, S, Bm, S, S, mb, nb, tid, sh, acc);
  int lane = tid & 31, wave = tid >> 5, wm = wave >> 1, wn = wave & 1;
  #pragma unroll
  for (int i = 0; i < 2; ++i) {
    int rb = mb + wm * 32 + i * 16 + (lane >> 4) * 8;
    #pragma unroll
    for (int j = 0; j < 4; ++j) {
      int col = nb + wn * 64 + j * 16 + (lane & 15);
      #pragma unroll
      for (int r = 0; r < 8; ++r)
        O[(size_t)(rb + r) * Dk + col] = acc[i][j][r];
    }
  }
}

// ---- Softmax over rows of length S; writes f16 probabilities ---------------
__global__ __launch_bounds__(256) void softmax_rows(const float* __restrict__ Sc,
                                                    half_t* __restrict__ P, int S) {
  const float* row = Sc + (size_t)blockIdx.x * S;
  half_t* out      = P  + (size_t)blockIdx.x * S;
  int tid = threadIdx.x;
  __shared__ float red[8];
  float lmax = -3.0e38f;
  for (int i = tid; i < S; i += 256) lmax = fmaxf(lmax, row[i]);
  #pragma unroll
  for (int o = 16; o; o >>= 1) lmax = fmaxf(lmax, __shfl_xor(lmax, o, 32));
  if ((tid & 31) == 0) red[tid >> 5] = lmax;
  __syncthreads();
  float m = red[0];
  #pragma unroll
  for (int j = 1; j < 8; ++j) m = fmaxf(m, red[j]);
  __syncthreads();
  float lsum = 0.f;
  for (int i = tid; i < S; i += 256) lsum += __expf(row[i] - m);
  #pragma unroll
  for (int o = 16; o; o >>= 1) lsum += __shfl_xor(lsum, o, 32);
  if ((tid & 31) == 0) red[tid >> 5] = lsum;
  __syncthreads();
  float s = 0.f;
  #pragma unroll
  for (int j = 0; j < 8; ++j) s += red[j];
  float inv = 1.0f / s;
  for (int i = tid; i < S; i += 256) out[i] = (half_t)(__expf(row[i] - m) * inv);
}

// ---- Converts --------------------------------------------------------------
__global__ __launch_bounds__(256) void f32_to_f16(const float* __restrict__ in,
                                                  half_t* __restrict__ out, int n) {
  int i = blockIdx.x * 256 + threadIdx.x;
  if (i < n) out[i] = (half_t)in[i];
}

// in [Kd][Nd] f32 -> out [Nd][Kd] f16
__global__ __launch_bounds__(256) void transpose_f32_to_f16(const float* __restrict__ in,
                                                            half_t* __restrict__ out,
                                                            int Kd, int Nd) {
  int i = blockIdx.x * 256 + threadIdx.x;
  if (i < Kd * Nd) {
    int k = i / Nd, n = i - k * Nd;
    out[(size_t)n * Kd + k] = (half_t)in[i];
  }
}

extern "C" void kernel_launch(void* const* d_in, const int* in_sizes, int n_in,
                              void* d_out, int out_size, void* d_ws, size_t ws_size,
                              hipStream_t stream) {
  (void)in_sizes; (void)n_in; (void)out_size; (void)ws_size;
  const float* x    = (const float*)d_in[0];
  const float* mask = (const float*)d_in[1];
  const float* Wq   = (const float*)d_in[2];
  const float* bq   = (const float*)d_in[3];
  const float* Wk   = (const float*)d_in[4];
  const float* bk   = (const float*)d_in[5];
  const float* Wv   = (const float*)d_in[6];
  const float* bv   = (const float*)d_in[7];
  float* out = (float*)d_out;

  const int Bb = 4, S = 2048, D = 1024, Dk = 1024;
  const int M = Bb * S;                      // 8192 merged (batch, seq) rows
  const float scale = 1.0f / 32.0f;          // 1/sqrt(1024)

  char* ws = (char*)d_ws;
  size_t off = 0;
  half_t* x_h   = (half_t*)(ws + off); off += (size_t)M * D * sizeof(half_t);
  half_t* Wq_t  = (half_t*)(ws + off); off += (size_t)D * Dk * sizeof(half_t);
  half_t* Wk_t  = (half_t*)(ws + off); off += (size_t)D * Dk * sizeof(half_t);
  half_t* Wv_t  = (half_t*)(ws + off); off += (size_t)D * Dk * sizeof(half_t);
  half_t* Qh    = (half_t*)(ws + off); off += (size_t)M * Dk * sizeof(half_t);
  half_t* Kh    = (half_t*)(ws + off); off += (size_t)M * Dk * sizeof(half_t);
  half_t* Vt    = (half_t*)(ws + off); off += (size_t)M * Dk * sizeof(half_t);  // [b][Dk][S]
  float*  Sc    = (float*)(ws + off);  off += (size_t)Bb * S * S * sizeof(float);
  half_t* P     = (half_t*)(ws + off); off += (size_t)Bb * S * S * sizeof(half_t);

  // 1) converts: x -> f16; W -> f16 transposed [Dk][D]
  int nx = M * D;
  f32_to_f16<<<(nx + 255) / 256, 256, 0, stream>>>(x, x_h, nx);
  int nw = D * Dk;
  transpose_f32_to_f16<<<(nw + 255) / 256, 256, 0, stream>>>(Wq, Wq_t, D, Dk);
  transpose_f32_to_f16<<<(nw + 255) / 256, 256, 0, stream>>>(Wk, Wk_t, D, Dk);
  transpose_f32_to_f16<<<(nw + 255) / 256, 256, 0, stream>>>(Wv, Wv_t, D, Dk);

  // 2) Q/K/V projections (WMMA, async double-buffered LDS staging)
  dim3 blk(256);
  dim3 g1(Dk / BN, M / BM);                  // (8, 64)
  gemm_bias<false><<<g1, blk, 0, stream>>>(x_h, Wq_t, bq, Qh, M, Dk, D, S);
  gemm_bias<false><<<g1, blk, 0, stream>>>(x_h, Wk_t, bk, Kh, M, Dk, D, S);
  gemm_bias<true ><<<g1, blk, 0, stream>>>(x_h, Wv_t, bv, Vt, M, Dk, D, S);  // V stored [b][Dk][S]

  // 3) scores = Q @ K^T * scale + mask
  dim3 g2(S / BN, S / BM, Bb);               // (16, 16, 4)
  gemm_scores<<<g2, blk, 0, stream>>>(Qh, Kh, mask, Sc, S, Dk, scale);

  // 4) row softmax -> P (f16)
  softmax_rows<<<Bb * S, blk, 0, stream>>>(Sc, P, S);

  // 5) out = P @ V
  dim3 g3(Dk / BN, S / BM, Bb);              // (8, 16, 4)
  gemm_pv<<<g3, blk, 0, stream>>>(P, Vt, out, S, Dk);
}
